// Model_39676907880957
// MI455X (gfx1250) — matrix operand order repaired and checked by reading, not compiled, not run on hardware
//
#include <hip/hip_runtime.h>
#include <hip/hip_bf16.h>

// ---------------------------------------------------------------------------
// Problem constants (match reference)
// ---------------------------------------------------------------------------
constexpr int Bc  = 2;
constexpr int Sc  = 2048;
constexpr int Dc  = 2048;
constexpr int Hc  = 8;
constexpr int DKc = Dc / Hc;          // 256
constexpr float SCALING = 0.0625f;    // DK^-0.5
constexpr float NEGBIG  = -1e9f;

typedef __attribute__((ext_vector_type(16))) __bf16 v16bf;
typedef __attribute__((ext_vector_type(8)))  __bf16 v8bf;
typedef __attribute__((ext_vector_type(8)))  float  v8f;

// ---------------------------------------------------------------------------
// gfx1250 async global->LDS path (ASYNCcnt-tracked). The builtin takes
// (int32x4 __device__* src, int32x4 __shared__* dst, imm offset, imm cpol).
// Detected at compile time; falls back to VGPR round-trip staging if absent.
// ---------------------------------------------------------------------------
#if defined(__has_builtin)
#if __has_builtin(__builtin_amdgcn_global_load_async_to_lds_b128) && \
    __has_builtin(__builtin_amdgcn_s_wait_asynccnt)
#define USE_ASYNC_LDS 1
#endif
#endif
#ifndef USE_ASYNC_LDS
#define USE_ASYNC_LDS 0
#endif

#if USE_ASYNC_LDS
typedef int b128_t __attribute__((vector_size(16)));
// LDS generic pointers carry the 32-bit LDS byte offset in addr[31:0]
// (ISA 10.2 aperture rules), so integer round-trips build the AS pointers.
#define ASYNC_CP16(dst_lds, src_glb)                                              \
    __builtin_amdgcn_global_load_async_to_lds_b128(                               \
        (__attribute__((address_space(1))) b128_t*)(uintptr_t)(src_glb),          \
        (__attribute__((address_space(3))) b128_t*)(uint32_t)(uintptr_t)(dst_lds),\
        0, 0)
__device__ __forceinline__ void async_publish_wait() {
    __builtin_amdgcn_s_wait_asynccnt(0);
}
#else
__device__ __forceinline__ void async_publish_wait() {}
#endif

__device__ __forceinline__ __bf16 tobf(float f) {
    unsigned u = __builtin_bit_cast(unsigned, f);
    u += 0x7FFFu + ((u >> 16) & 1u);              // round-to-nearest-even
    unsigned short h = (unsigned short)(u >> 16);
    return __builtin_bit_cast(__bf16, h);
}

__device__ __forceinline__ v16bf cat16(v8bf lo, v8bf hi) {
    return __builtin_shufflevector(lo, hi, 0,1,2,3,4,5,6,7,8,9,10,11,12,13,14,15);
}

// A-operand of v_wmma_f32_16x16x32_bf16: per-lane element e -> K = 8*half + e
// + (e>=8 ? 8 : 0)  => two contiguous 16B chunks at +8*half and +16+8*half.
__device__ __forceinline__ v16bf load_a_tile(const __bf16* rowp, int half) {
    v8bf lo = *(const v8bf*)(rowp + half * 8);
    v8bf hi = *(const v8bf*)(rowp + 16 + half * 8);
    return cat16(lo, hi);
}
// B column gather: element e -> K = 8*half + e + (e>=8 ? 8 : 0), row stride `stride`
__device__ __forceinline__ v16bf load16(const __bf16* p, int stride, int half) {
    v16bf r;
    #pragma unroll
    for (int e = 0; e < 16; ++e)
        r[e] = p[(8 * half + e + (e >= 8 ? 8 : 0)) * stride];
    return r;
}

// ---------------------------------------------------------------------------
// Kernel 1: f32 -> bf16 conversion of x
// ---------------------------------------------------------------------------
__global__ __launch_bounds__(256) void xconv_kernel(const float* __restrict__ x,
                                                    __bf16* __restrict__ xb, int n) {
    int i = blockIdx.x * 256 + threadIdx.x;
    if (i < n) xb[i] = tobf(x[i]);
}

// ---------------------------------------------------------------------------
// Kernel 2: transpose + convert Wq/Wk/Wv into WT[m][k][n] bf16 (LDS 32x32 tiles)
// ---------------------------------------------------------------------------
__global__ __launch_bounds__(256) void wt_kernel(const float* __restrict__ Wq,
                                                 const float* __restrict__ Wk,
                                                 const float* __restrict__ Wv,
                                                 __bf16* __restrict__ WT) {
    __shared__ __bf16 tile[32][33];
    const int m = blockIdx.z;
    const float* W = (m == 0) ? Wq : (m == 1) ? Wk : Wv;
    const int n0 = blockIdx.x * 32;
    const int k0 = blockIdx.y * 32;
    for (int i = threadIdx.y; i < 32; i += 8)
        tile[i][threadIdx.x] = tobf(W[(size_t)(n0 + i) * Dc + k0 + threadIdx.x]);
    __syncthreads();
    __bf16* out = WT + (size_t)m * Dc * Dc;
    for (int i = threadIdx.y; i < 32; i += 8)
        out[(size_t)(k0 + i) * Dc + n0 + threadIdx.x] = tile[threadIdx.x][i];
}

// ---------------------------------------------------------------------------
// Kernel 3: LDS-tiled QKV GEMM.  Block (256 thr / 8 waves) owns a 128x128
// output tile of one of {q,k,v}; per K-step of 32 the A (128x32) and B
// (32x128) slabs are double-buffered through LDS (async if available).
// Wave w: rows 32*(w&3), cols 64*(w>>2) -> 2x4 = 8 WMMA accumulators.
// ---------------------------------------------------------------------------
constexpr int ASTR = 40;    // A LDS row stride (elems): 80B, 16B aligned, bank-staggered
constexpr int BSTR = 136;   // B LDS row stride (elems): 272B
constexpr int KSTEPS = Dc / 32;

__global__ __launch_bounds__(256) void qkv_kernel(const __bf16* __restrict__ xb,
                                                  const __bf16* __restrict__ WT,
                                                  const float* __restrict__ bq,
                                                  const float* __restrict__ bk,
                                                  const float* __restrict__ bv,
                                                  __bf16* __restrict__ qb,
                                                  __bf16* __restrict__ kT,
                                                  float* __restrict__ vf) {
    __shared__ __bf16 Abuf[2][128 * ASTR];
    __shared__ __bf16 Bbuf[2][32 * BSTR];

    const int sel  = blockIdx.z;                 // 0=q 1=k 2=v
    const int tid  = threadIdx.x;
    const int wave = tid >> 5;
    const int lane = tid & 31;
    const int row  = lane & 15;
    const int half = lane >> 4;
    const int wm   = wave & 3;                   // row slab  (32 rows)
    const int wn   = wave >> 2;                  // col slab  (64 cols)

    const int n0 = blockIdx.x * 128;
    const int m0 = blockIdx.y * 128;

    const __bf16* W    = WT + (size_t)sel * Dc * Dc;
    const float*  bias = (sel == 0) ? bq : (sel == 1) ? bk : bv;

    // staging map: A: thread t -> row t>>1, 32B chunk (t&1); B: row t>>3, chunk t&7
    const int arow_s = tid >> 1, acol_s = (tid & 1) * 16;
    const int brow_s = tid >> 3, bcol_s = (tid & 7) * 16;
    const __bf16* gA = xb + (size_t)(m0 + arow_s) * Dc + acol_s;
    const __bf16* gB = W + (size_t)brow_s * Dc + n0 + bcol_s;
    __bf16* lA0 = &Abuf[0][arow_s * ASTR + acol_s];
    __bf16* lB0 = &Bbuf[0][brow_s * BSTR + bcol_s];
    __bf16* lA1 = &Abuf[1][arow_s * ASTR + acol_s];
    __bf16* lB1 = &Bbuf[1][brow_s * BSTR + bcol_s];

#if !USE_ASYNC_LDS
    v8bf ra0, ra1, rb0, rb1;
#endif

    // ---- prologue: stage k-step 0 into buffer 0
#if USE_ASYNC_LDS
    ASYNC_CP16(lA0, gA);           ASYNC_CP16(lA0 + 8, gA + 8);
    ASYNC_CP16(lB0, gB);           ASYNC_CP16(lB0 + 8, gB + 8);
#else
    ra0 = *(const v8bf*)gA;        ra1 = *(const v8bf*)(gA + 8);
    rb0 = *(const v8bf*)gB;        rb1 = *(const v8bf*)(gB + 8);
    *(v8bf*)lA0 = ra0;             *(v8bf*)(lA0 + 8) = ra1;
    *(v8bf*)lB0 = rb0;             *(v8bf*)(lB0 + 8) = rb1;
#endif
    async_publish_wait();
    __syncthreads();

    v8f z = {};
    v8f acc[8] = {z, z, z, z, z, z, z, z};       // [tm*4 + tn]

    for (int ks = 0; ks < KSTEPS; ++ks) {
        const int cur = ks & 1;
        const bool more = (ks + 1) < KSTEPS;
        const __bf16* gAn = gA + (ks + 1) * 32;
        const __bf16* gBn = gB + (size_t)(ks + 1) * 32 * Dc;
        __bf16* lAn = cur ? lA0 : lA1;
        __bf16* lBn = cur ? lB0 : lB1;

        // issue next-step staging (overlaps with WMMA below)
        if (more) {
#if USE_ASYNC_LDS
            ASYNC_CP16(lAn, gAn);  ASYNC_CP16(lAn + 8, gAn + 8);
            ASYNC_CP16(lBn, gBn);  ASYNC_CP16(lBn + 8, gBn + 8);
#else
            ra0 = *(const v8bf*)gAn;  ra1 = *(const v8bf*)(gAn + 8);
            rb0 = *(const v8bf*)gBn;  rb1 = *(const v8bf*)(gBn + 8);
#endif
        }

        // compute current step from LDS
        const __bf16* Ab = Abuf[cur];
        const __bf16* Bb = Bbuf[cur];
        v16bf a0 = load_a_tile(Ab + (wm * 32 + row) * ASTR, half);
        v16bf a1 = load_a_tile(Ab + (wm * 32 + 16 + row) * ASTR, half);
        const __bf16* brow_p = Bb + row + wn * 64;
        v16bf b0 = load16(brow_p, BSTR, half);
        v16bf b1 = load16(brow_p + 16, BSTR, half);
        v16bf b2 = load16(brow_p + 32, BSTR, half);
        v16bf b3 = load16(brow_p + 48, BSTR, half);
        acc[0] = __builtin_amdgcn_wmma_f32_16x16x32_bf16(false, a0, false, b0, (short)0, acc[0], false, false);
        acc[1] = __builtin_amdgcn_wmma_f32_16x16x32_bf16(false, a0, false, b1, (short)0, acc[1], false, false);
        acc[2] = __builtin_amdgcn_wmma_f32_16x16x32_bf16(false, a0, false, b2, (short)0, acc[2], false, false);
        acc[3] = __builtin_amdgcn_wmma_f32_16x16x32_bf16(false, a0, false, b3, (short)0, acc[3], false, false);
        acc[4] = __builtin_amdgcn_wmma_f32_16x16x32_bf16(false, a1, false, b0, (short)0, acc[4], false, false);
        acc[5] = __builtin_amdgcn_wmma_f32_16x16x32_bf16(false, a1, false, b1, (short)0, acc[5], false, false);
        acc[6] = __builtin_amdgcn_wmma_f32_16x16x32_bf16(false, a1, false, b2, (short)0, acc[6], false, false);
        acc[7] = __builtin_amdgcn_wmma_f32_16x16x32_bf16(false, a1, false, b3, (short)0, acc[7], false, false);

        if (more) {
#if !USE_ASYNC_LDS
            *(v8bf*)lAn = ra0;  *(v8bf*)(lAn + 8) = ra1;
            *(v8bf*)lBn = rb0;  *(v8bf*)(lBn + 8) = rb1;
#endif
            async_publish_wait();
        }
        __syncthreads();
    }

    // ---- epilogue: C/D layout lane(row,half), reg r -> [M=r+8*half][N=row]
    #pragma unroll
    for (int tm = 0; tm < 2; ++tm) {
        #pragma unroll
        for (int tn = 0; tn < 4; ++tn) {
            const v8f& a = acc[tm * 4 + tn];
            const int nb = n0 + wn * 64 + tn * 16 + row;
            const float bs = bias[nb];
            #pragma unroll
            for (int r = 0; r < 8; ++r) {
                const float val = a[r] + bs;
                const int gs = m0 + wm * 32 + tm * 16 + r + 8 * half;
                if (sel == 0) {
                    qb[(size_t)gs * Dc + nb] = tobf(val);
                } else if (sel == 2) {
                    vf[(size_t)gs * Dc + nb] = val;
                } else {
                    const int b = gs / Sc, srow = gs % Sc;
                    const int h = nb / DKc, dk = nb % DKc;
                    kT[(((size_t)b * Hc + h) * DKc + dk) * Sc + srow] = tobf(val);
                }
            }
        }
    }
}

// ---------------------------------------------------------------------------
// Kernel 4: fused attention score with LDS-shared key tiles.
// Block (8 waves) owns 8 query tiles of one (b,h); the 256x16 kT slab per
// 16-key step is double-buffered through LDS and consumed by all 8 waves.
//   score[b,h,i] = SCALING*( sum_j leaky_relu(q_i . k_j)*Wa[j] + ba )
// ---------------------------------------------------------------------------
constexpr int KSTR = 24;    // kT LDS row stride (elems): 48B, 16B aligned

__global__ __launch_bounds__(256) void score_kernel(const __bf16* __restrict__ qb,
                                                    const __bf16* __restrict__ kT,
                                                    const float* __restrict__ Wa,
                                                    const float* __restrict__ ba,
                                                    float* __restrict__ score) {
    __shared__ __bf16 Kb[2][256 * KSTR];

    const int tid  = threadIdx.x;
    const int wave = tid >> 5;
    const int lane = tid & 31;
    const int row  = lane & 15;
    const int half = lane >> 4;

    const int bh = blockIdx.x >> 4;              // 16 blocks per (b,h)
    const int it = (blockIdx.x & 15) * 8 + wave; // query tile 0..127
    const int b = bh / Hc, h = bh % Hc;
    const int i0 = it * 16;

    // Preload entire q A-tile (16 queries x 256 dk) into registers
    const __bf16* qrow = qb + (size_t)(b * Sc + i0 + row) * Dc + h * DKc;
    v16bf aregs[8];
    #pragma unroll
    for (int dkt = 0; dkt < 8; ++dkt)
        aregs[dkt] = load_a_tile(qrow + dkt * 32, half);

    // staging map: thread t loads kT row dk=t (16 j's = 32B)
    const __bf16* gK = kT + ((size_t)bh * DKc + tid) * Sc;
    __bf16* lK0 = &Kb[0][tid * KSTR];
    __bf16* lK1 = &Kb[1][tid * KSTR];

#if !USE_ASYNC_LDS
    v8bf rk0, rk1;
#endif

    // prologue: stage jt=0 into buffer 0
#if USE_ASYNC_LDS
    ASYNC_CP16(lK0, gK);  ASYNC_CP16(lK0 + 8, gK + 8);
#else
    rk0 = *(const v8bf*)gK;  rk1 = *(const v8bf*)(gK + 8);
    *(v8bf*)lK0 = rk0;       *(v8bf*)(lK0 + 8) = rk1;
#endif
    async_publish_wait();
    __syncthreads();

    float srow[8] = {0.f, 0.f, 0.f, 0.f, 0.f, 0.f, 0.f, 0.f};

    for (int js = 0; js < Sc / 16; ++js) {
        const int cur = js & 1;
        const bool more = (js + 1) < (Sc / 16);
        const __bf16* gKn = gK + (js + 1) * 16;
        __bf16* lKn = cur ? lK0 : lK1;

        if (more) {
#if USE_ASYNC_LDS
            ASYNC_CP16(lKn, gKn);  ASYNC_CP16(lKn + 8, gKn + 8);
#else
            rk0 = *(const v8bf*)gKn;  rk1 = *(const v8bf*)(gKn + 8);
#endif
        }

        const __bf16* Kc = Kb[cur];
        v8f acc = {};
        #pragma unroll
        for (int dkt = 0; dkt < 8; ++dkt) {
            v16bf bb = load16(Kc + dkt * 32 * KSTR + row, KSTR, half);
            acc = __builtin_amdgcn_wmma_f32_16x16x32_bf16(false, aregs[dkt], false, bb,
                                                          (short)0, acc, false, false);
        }
        const float wa = Wa[js * 16 + row];      // this lane's key column
        #pragma unroll
        for (int r = 0; r < 8; ++r) {
            float e = acc[r];
            e = (e > 0.f) ? e : 0.1f * e;        // leaky_relu(., 0.1)
            srow[r] += e * wa;
        }

        if (more) {
#if !USE_ASYNC_LDS
            *(v8bf*)lKn = rk0;  *(v8bf*)(lKn + 8) = rk1;
#endif
            async_publish_wait();
        }
        __syncthreads();
    }

    // reduce over the 16 N-lanes of each wave half (masks < 16 stay in-half)
    const float bav = ba[0];
    #pragma unroll
    for (int r = 0; r < 8; ++r) {
        float v = srow[r];
        v += __shfl_xor(v, 1);
        v += __shfl_xor(v, 2);
        v += __shfl_xor(v, 4);
        v += __shfl_xor(v, 8);
        if (row == 0)
            score[(size_t)bh * Sc + i0 + r + 8 * half] = SCALING * (v + bav);
    }
}

// ---------------------------------------------------------------------------
// Kernel 5: masked softmax over the token axis, one block per (b,h)
// ---------------------------------------------------------------------------
__global__ __launch_bounds__(256) void softmax_kernel(const float* __restrict__ score,
                                                      const int* __restrict__ x_mask,
                                                      float* __restrict__ attn) {
    const int bh = blockIdx.x;
    const float* s = score + (size_t)bh * Sc;
    float* a = attn + (size_t)bh * Sc;
    __shared__ float red[256];

    float vals[8];
    float m = -3.4e38f;
    #pragma unroll
    for (int k = 0; k < 8; ++k) {
        const int i = threadIdx.x + k * 256;
        float v = s[i];
        if (x_mask[i] != 0) v = NEGBIG;
        vals[k] = v;
        m = fmaxf(m, v);
    }
    red[threadIdx.x] = m;
    __syncthreads();
    for (int off = 128; off > 0; off >>= 1) {
        if (threadIdx.x < off) red[threadIdx.x] = fmaxf(red[threadIdx.x], red[threadIdx.x + off]);
        __syncthreads();
    }
    m = red[0];
    __syncthreads();

    float sum = 0.f;
    #pragma unroll
    for (int k = 0; k < 8; ++k) {
        vals[k] = __expf(vals[k] - m);
        sum += vals[k];
    }
    red[threadIdx.x] = sum;
    __syncthreads();
    for (int off = 128; off > 0; off >>= 1) {
        if (threadIdx.x < off) red[threadIdx.x] += red[threadIdx.x + off];
        __syncthreads();
    }
    const float inv = 1.0f / red[0];
    #pragma unroll
    for (int k = 0; k < 8; ++k)
        a[threadIdx.x + k * 256] = vals[k] * inv;
}

// ---------------------------------------------------------------------------
// Kernel 6: attention-weighted pooling
// ---------------------------------------------------------------------------
__global__ __launch_bounds__(256) void pool_kernel(const float* __restrict__ attn,
                                                   const float* __restrict__ vf,
                                                   float* __restrict__ pooled) {
    const int idx = blockIdx.x * 256 + threadIdx.x;   // 0 .. B*D-1
    const int b = idx / Dc, d = idx % Dc;
    const int h = d / DKc;
    const float* arow = attn + (size_t)(b * Hc + h) * Sc;
    const float* vcol = vf + (size_t)b * Sc * Dc + d;
    float acc = 0.f;
    for (int s = 0; s < Sc; ++s)
        acc = fmaf(arow[s], vcol[(size_t)s * Dc], acc);
    pooled[idx] = acc;
}

// ---------------------------------------------------------------------------
// Kernel 7: final projection  out = tanh(pooled @ Wp.T + bp)
// ---------------------------------------------------------------------------
__global__ __launch_bounds__(256) void final_kernel(const float* __restrict__ pooled,
                                                    const float* __restrict__ Wp,
                                                    const float* __restrict__ bp,
                                                    float* __restrict__ out) {
    const int idx = blockIdx.x * 256 + threadIdx.x;   // 0 .. B*D-1
    const int b = idx / Dc, n = idx % Dc;
    const float* p = pooled + (size_t)b * Dc;
    const float* w = Wp + (size_t)n * Dc;
    float acc = bp[n];
    for (int d = 0; d < Dc; ++d)
        acc = fmaf(p[d], w[d], acc);
    out[idx] = tanhf(acc);
}

// ---------------------------------------------------------------------------
// Launch
// ---------------------------------------------------------------------------
extern "C" void kernel_launch(void* const* d_in, const int* in_sizes, int n_in,
                              void* d_out, int out_size, void* d_ws, size_t ws_size,
                              hipStream_t stream) {
    const float* x      = (const float*)d_in[0];
    const int*   x_mask = (const int*)  d_in[1];
    const float* Wq = (const float*)d_in[2];  const float* bq = (const float*)d_in[3];
    const float* Wk = (const float*)d_in[4];  const float* bk = (const float*)d_in[5];
    const float* Wv = (const float*)d_in[6];  const float* bv = (const float*)d_in[7];
    const float* Wa = (const float*)d_in[8];  const float* ba = (const float*)d_in[9];
    const float* Wp = (const float*)d_in[10]; const float* bp = (const float*)d_in[11];
    float* out = (float*)d_out;

    auto alignup = [](size_t v) { return (v + 255) & ~(size_t)255; };
    char* ws = (char*)d_ws;
    size_t off = 0;
    __bf16* xb  = (__bf16*)(ws + off); off += alignup((size_t)Bc * Sc * Dc * 2);
    __bf16* WT  = (__bf16*)(ws + off); off += alignup((size_t)3 * Dc * Dc * 2);
    __bf16* qb  = (__bf16*)(ws + off); off += alignup((size_t)Bc * Sc * Dc * 2);
    __bf16* kT  = (__bf16*)(ws + off); off += alignup((size_t)Bc * Sc * Dc * 2);
    float*  vf  = (float*) (ws + off); off += alignup((size_t)Bc * Sc * Dc * 4);
    float*  sc  = (float*) (ws + off); off += alignup((size_t)Bc * Hc * Sc * 4);
    float*  at  = (float*) (ws + off); off += alignup((size_t)Bc * Hc * Sc * 4);
    float*  pl  = (float*) (ws + off); off += alignup((size_t)Bc * Dc * 4);
    (void)ws_size; (void)in_sizes; (void)n_in; (void)out_size;

    // 1) x -> bf16
    {
        const int n = Bc * Sc * Dc;
        xconv_kernel<<<n / 256, 256, 0, stream>>>(x, xb, n);
    }
    // 2) Wq/Wk/Wv -> transposed bf16
    {
        dim3 grid(Dc / 32, Dc / 32, 3);
        wt_kernel<<<grid, dim3(32, 8), 0, stream>>>(Wq, Wk, Wv, WT);
    }
    // 3) LDS-tiled QKV WMMA GEMM: 128x128 tile per block
    {
        dim3 grid(Dc / 128, (Bc * Sc) / 128, 3);
        qkv_kernel<<<grid, 256, 0, stream>>>(xb, WT, bq, bk, bv, qb, kT, vf);
    }
    // 4) fused attention score with LDS-shared key tiles
    {
        score_kernel<<<256, 256, 0, stream>>>(qb, kT, Wa, ba, sc);
    }
    // 5) masked softmax per (b,h)
    {
        softmax_kernel<<<Bc * Hc, 256, 0, stream>>>(sc, x_mask, at);
    }
    // 6) pooled = attn-weighted sum of v
    {
        pool_kernel<<<(Bc * Dc) / 256, 256, 0, stream>>>(at, vf, pl);
    }
    // 7) out = tanh(pooled @ Wp.T + bp)
    {
        final_kernel<<<(Bc * Dc) / 256, 256, 0, stream>>>(pl, Wp, bp, out);
    }
}
